// GCN_Encoder_27754078666900
// MI455X (gfx1250) — matrix operand order repaired and checked by reading, not compiled, not run on hardware
//
#include <hip/hip_runtime.h>
#include <hip/hip_bf16.h>
#include <math.h>

typedef __attribute__((ext_vector_type(16))) __bf16 v16bf;
typedef __attribute__((ext_vector_type(8)))  float  v8f;
typedef __attribute__((ext_vector_type(4)))  float  f4v;

// ---------------------------------------------------------------------------
// Degree / normalization kernels
// ---------------------------------------------------------------------------
__global__ void k_init_deg(float* __restrict__ deg, int N) {
    int i = blockIdx.x * blockDim.x + threadIdx.x;
    if (i < N) deg[i] = 1.0f;  // self-loop contributes 1 to every degree
}

__global__ void k_deg_count(const long long* __restrict__ ei, int E,
                            float* __restrict__ deg) {
    int e = blockIdx.x * blockDim.x + threadIdx.x;
    if (e < E) atomicAdd(&deg[(int)ei[(long long)E + e]], 1.0f);
}

__global__ void k_make_dinv(float* __restrict__ deg, int N) {
    int i = blockIdx.x * blockDim.x + threadIdx.x;
    if (i < N) {
        float d = deg[i];
        deg[i] = (d > 0.0f) ? rsqrtf(d) : 0.0f;  // in place: deg -> dinv
    }
}

// ---------------------------------------------------------------------------
// GEMM: H[N x C] = X[N x K] @ W[K x C], bf16 WMMA with f32 accumulation.
//
// W staged in LDS pre-permuted into B-fragment order (one fragment = 32
// lanes x 16 bf16 contiguous per lane -> fragment load = 2x ds_load_b128).
// Each wave computes TWO 16-row tiles (32 rows): every B fragment feeds two
// back-to-back WMMAs, and B fragments are double-buffered so the ds latency
// overlaps the in-flight WMMAs. 256 threads = 8 waves -> 256 rows per block.
// Full-tile blocks (all but the last) take an unpredicated store fast path.
// ---------------------------------------------------------------------------
template <int K, int C>
__global__ __launch_bounds__(256, 1) void k_gemm_bf16(
    const float* __restrict__ X, const float* __restrict__ W,
    float* __restrict__ H, int N)
{
    extern __shared__ __attribute__((aligned(32))) __bf16 sW[];  // K*C bf16
    v16bf* __restrict__ sWv = (v16bf*)sW;          // K*C/16 fragments-of-16

    constexpr int NT = C / 16;         // column tiles
    constexpr int NF = (K / 32) * NT;  // total B fragments

    // Stage + permute W into fragment order (one-time, 2x ds_store_b128 per
    // 16 strided W reads).
    for (int p = threadIdx.x; p < NF * 32; p += blockDim.x) {
        const int lane = p & 31;
        const int f    = p >> 5;
        const int t    = f % NT;
        const int kblk = f / NT;
        const int c    = t * 16 + (lane & 15);
        const int kb   = kblk * 32 + ((lane & 16) ? 8 : 0);
        v16bf v;
#pragma unroll
        for (int e = 0; e < 16; ++e) v[e] = (__bf16)W[(kb + e + ((e >= 8) ? 8 : 0)) * C + c];
        sWv[p] = v;
    }
    __syncthreads();

    const int lane    = threadIdx.x & 31;
    const int wv      = threadIdx.x >> 5;
    const int rowBase = blockIdx.x * 256 + wv * 32;  // 32 rows per wave
    const int m       = lane & 15;
    const bool hi     = lane >= 16;
    const bool full   = (rowBase + 32) <= N;         // uniform per wave

    long long r0c = rowBase + m;
    long long r1c = rowBase + 16 + m;
    if (!full) {  // clamp only in the (rare) tail block
        if (r0c > N - 1) r0c = N - 1;
        if (r1c > N - 1) r1c = N - 1;
    }
    const float* xrow0 = X + r0c * K;
    const float* xrow1 = X + r1c * K;

    v8f acc0[NT], acc1[NT];
#pragma unroll
    for (int t = 0; t < NT; ++t) {
        acc0[t] = (v8f){0,0,0,0,0,0,0,0};
        acc1[t] = (v8f){0,0,0,0,0,0,0,0};
    }

#pragma unroll 1
    for (int k0 = 0; k0 < K; k0 += 32) {
        __builtin_prefetch(xrow0 + k0 + 32, 0, 1);  // global_prefetch_b8
        __builtin_prefetch(xrow1 + k0 + 32, 0, 1);

        // A fragments (16-bit A 16x32 layout), 4x global_load_b128 each.
        const int kb = k0 + (hi ? 8 : 0);
        const f4v a00 = *(const f4v*)(xrow0 + kb);
        const f4v a01 = *(const f4v*)(xrow0 + kb + 4);
        const f4v a02 = *(const f4v*)(xrow0 + kb + 16);
        const f4v a03 = *(const f4v*)(xrow0 + kb + 20);
        const f4v a10 = *(const f4v*)(xrow1 + kb);
        const f4v a11 = *(const f4v*)(xrow1 + kb + 4);
        const f4v a12 = *(const f4v*)(xrow1 + kb + 16);
        const f4v a13 = *(const f4v*)(xrow1 + kb + 20);

        v16bf a0, a1;
#pragma unroll
        for (int i = 0; i < 4; ++i) {
            a0[i]      = (__bf16)a00[i];
            a0[4 + i]  = (__bf16)a01[i];
            a0[8 + i]  = (__bf16)a02[i];
            a0[12 + i] = (__bf16)a03[i];
            a1[i]      = (__bf16)a10[i];
            a1[4 + i]  = (__bf16)a11[i];
            a1[8 + i]  = (__bf16)a12[i];
            a1[12 + i] = (__bf16)a13[i];
        }

        const int fbase = (k0 >> 5) * NT;
        v16bf bcur = sWv[fbase * 32 + lane];
#pragma unroll
        for (int t = 0; t < NT; ++t) {
            v16bf bnxt = bcur;
            if (t + 1 < NT) bnxt = sWv[(fbase + t + 1) * 32 + lane];
            acc0[t] = __builtin_amdgcn_wmma_f32_16x16x32_bf16(
                false, a0, false, bcur, (short)0, acc0[t], false, false);
            acc1[t] = __builtin_amdgcn_wmma_f32_16x16x32_bf16(
                false, a1, false, bcur, (short)0, acc1[t], false, false);
            bcur = bnxt;
        }
    }

    // C/D layout: VGPR v -> row tileBase + v + 8*hi, col = t*16 + (lane&15)
    const int s0 = rowBase + (hi ? 8 : 0);
    const int s1 = rowBase + 16 + (hi ? 8 : 0);
    if (full) {
        // fast path: unpredicated stores
        float* __restrict__ p0 = H + (long long)s0 * C + m;
        float* __restrict__ p1 = H + (long long)s1 * C + m;
#pragma unroll
        for (int t = 0; t < NT; ++t) {
#pragma unroll
            for (int v = 0; v < 8; ++v) {
                p0[(long long)v * C + t * 16] = acc0[t][v];
                p1[(long long)v * C + t * 16] = acc1[t][v];
            }
        }
    } else {
#pragma unroll
        for (int t = 0; t < NT; ++t) {
#pragma unroll
            for (int v = 0; v < 8; ++v) {
                int ra = s0 + v;
                int rb = s1 + v;
                if (ra < N) H[(long long)ra * C + t * 16 + m] = acc0[t][v];
                if (rb < N) H[(long long)rb * C + t * 16 + m] = acc1[t][v];
            }
        }
    }
}

// ---------------------------------------------------------------------------
// agg[i,:] = h[i,:] * dinv[i]^2   (self-loop term; initializes the buffer)
// ---------------------------------------------------------------------------
template <int C>
__global__ void k_self_init(const float* __restrict__ H,
                            const float* __restrict__ dinv,
                            float* __restrict__ out, int N)
{
    long long tid = (long long)blockIdx.x * blockDim.x + threadIdx.x;
    if (tid >= (long long)N * C) return;
    int r = (int)(tid / C);
    float d = dinv[r];
    out[tid] = H[tid] * d * d;
}

// ---------------------------------------------------------------------------
// Edge scatter: out[dst,:] += h[src,:] * dinv[src]*dinv[dst]
// One thread per (edge, 4-feature chunk); float4 gather + 4x f32 atomics.
// Working set (h + agg, layer 1 = 102MB) fits the 192MB L2, so the random
// gather/atomic traffic is served at L2, not HBM.
// ---------------------------------------------------------------------------
template <int C>
__global__ void k_edge_scatter(const long long* __restrict__ ei, int E,
                               const float* __restrict__ H,
                               const float* __restrict__ dinv,
                               float* __restrict__ out)
{
    const int CH = C / 4;
    long long tid = (long long)blockIdx.x * blockDim.x + threadIdx.x;
    if (tid >= (long long)E * CH) return;
    int e  = (int)(tid / CH);
    int c4 = (int)(tid % CH) * 4;

    long long src = ei[e];
    long long dst = ei[(long long)E + e];
    float nrm = dinv[src] * dinv[dst];

    const f4v hv = *(const f4v*)(H + src * C + c4);  // global_load_b128
    float* od = out + dst * C + c4;
    __builtin_prefetch(od, 1, 1);
#pragma unroll
    for (int i = 0; i < 4; ++i)
        atomicAdd(od + i, hv[i] * nrm);
}

// ---------------------------------------------------------------------------
// Epilogue 1: h = dropout(relu(agg + b1)), deterministic hash mask (p=0.1)
// ---------------------------------------------------------------------------
__global__ void k_post1(const float* __restrict__ agg,
                        const float* __restrict__ b,
                        float* __restrict__ out, int N, int C)
{
    long long tid = (long long)blockIdx.x * blockDim.x + threadIdx.x;
    if (tid >= (long long)N * C) return;
    int c = (int)(tid % C);
    float v = agg[tid] + b[c];
    v = v > 0.0f ? v : 0.0f;

    unsigned h = (unsigned)tid;
    h *= 2654435761u; h ^= h >> 16;
    h *= 2246822519u; h ^= h >> 13;
    h *= 3266489917u; h ^= h >> 16;
    bool keep = (h & 0xFFFFFFu) < 15099494u;  // 0.9 * 2^24
    out[tid] = keep ? v * (1.0f / 0.9f) : 0.0f;
}

// ---------------------------------------------------------------------------
// Epilogue 2: y = agg + b2; L2-normalize; log_softmax. One wave per row
// (C=64 -> 2 elems/lane), wave32 shuffle reductions.
// ---------------------------------------------------------------------------
__global__ __launch_bounds__(256) void k_finalize(
    const float* __restrict__ agg, const float* __restrict__ b,
    float* __restrict__ out, int N)
{
    const int C = 64;
    int lane = threadIdx.x & 31;
    int wv   = threadIdx.x >> 5;
    int row  = blockIdx.x * 8 + wv;
    if (row >= N) return;

    const float* a = agg + (long long)row * C;
    float v0 = a[lane]      + b[lane];
    float v1 = a[lane + 32] + b[lane + 32];

    float ss = v0 * v0 + v1 * v1;
#pragma unroll
    for (int off = 16; off > 0; off >>= 1) ss += __shfl_xor(ss, off, 32);
    float inv = 1.0f / fmaxf(sqrtf(ss), 1e-12f);
    v0 *= inv; v1 *= inv;

    float mx = fmaxf(v0, v1);
#pragma unroll
    for (int off = 16; off > 0; off >>= 1) mx = fmaxf(mx, __shfl_xor(mx, off, 32));
    float es = __expf(v0 - mx) + __expf(v1 - mx);
#pragma unroll
    for (int off = 16; off > 0; off >>= 1) es += __shfl_xor(es, off, 32);
    float lse = __logf(es) + mx;

    out[(long long)row * C + lane]      = v0 - lse;
    out[(long long)row * C + lane + 32] = v1 - lse;
}

// ---------------------------------------------------------------------------
// Host orchestration
// ---------------------------------------------------------------------------
extern "C" void kernel_launch(void* const* d_in, const int* in_sizes, int n_in,
                              void* d_out, int out_size, void* d_ws, size_t ws_size,
                              hipStream_t stream)
{
    const float*     x  = (const float*)d_in[0];       // [N, 256]
    const float*     W1 = (const float*)d_in[1];       // [256, 128]
    const float*     b1 = (const float*)d_in[2];       // [128]
    const float*     W2 = (const float*)d_in[3];       // [128, 64]
    const float*     b2 = (const float*)d_in[4];       // [64]
    const long long* ei = (const long long*)d_in[5];   // [2, E]

    const int N = in_sizes[0] / 256;
    const int E = in_sizes[5] / 2;
    const int Chid = 128, Cout = 64;

    // Workspace carve-up (floats)
    float* deg  = (float*)d_ws;                         // N   (becomes dinv)
    float* h1   = deg  + N;                             // N*128
    float* agg1 = h1   + (size_t)N * Chid;              // N*128
    float* h2   = agg1 + (size_t)N * Chid;              // N*64
    float* agg2 = h2   + (size_t)N * Cout;              // N*64
    float* dinv = deg;
    float* outp = (float*)d_out;

    const int TB = 256;

    // 1) degrees -> dinv
    k_init_deg <<<(N + TB - 1) / TB, TB, 0, stream>>>(deg, N);
    k_deg_count<<<(E + TB - 1) / TB, TB, 0, stream>>>(ei, E, deg);
    k_make_dinv<<<(N + TB - 1) / TB, TB, 0, stream>>>(deg, N);

    // 2) layer 1: GEMM -> self-loop init -> edge scatter -> relu/dropout
    {
        int grid = (N + 255) / 256;
        size_t shmem = (size_t)256 * 128 * sizeof(__bf16);  // 64KB
        k_gemm_bf16<256, 128><<<grid, TB, shmem, stream>>>(x, W1, h1, N);
    }
    {
        long long tot = (long long)N * Chid;
        k_self_init<128><<<(int)((tot + TB - 1) / TB), TB, 0, stream>>>(h1, dinv, agg1, N);
    }
    {
        long long tot = (long long)E * (Chid / 4);
        k_edge_scatter<128><<<(int)((tot + TB - 1) / TB), TB, 0, stream>>>(ei, E, h1, dinv, agg1);
    }
    {
        long long tot = (long long)N * Chid;
        k_post1<<<(int)((tot + TB - 1) / TB), TB, 0, stream>>>(agg1, b1, h1, N, Chid);  // reuse h1
    }

    // 3) layer 2: GEMM -> self-loop init -> edge scatter
    {
        int grid = (N + 255) / 256;
        size_t shmem = (size_t)128 * 64 * sizeof(__bf16);   // 16KB
        k_gemm_bf16<128, 64><<<grid, TB, shmem, stream>>>(h1, W2, h2, N);
    }
    {
        long long tot = (long long)N * Cout;
        k_self_init<64><<<(int)((tot + TB - 1) / TB), TB, 0, stream>>>(h2, dinv, agg2, N);
    }
    {
        long long tot = (long long)E * (Cout / 4);
        k_edge_scatter<64><<<(int)((tot + TB - 1) / TB), TB, 0, stream>>>(ei, E, h2, dinv, agg2);
    }

    // 4) bias + L2 normalize + log_softmax
    k_finalize<<<(N + 7) / 8, TB, 0, stream>>>(agg2, b2, outp, N);
}
